// IPAdapterAttnProcessor_11596411699931
// MI455X (gfx1250) — matrix operand order repaired and checked by reading, not compiled, not run on hardware
//
#include <hip/hip_runtime.h>
#include <hip/hip_bf16.h>
#include <math.h>

// ---------------- problem constants ----------------
#define B_    4
#define S_    4096
#define STXT_ 77
#define C_    1280
#define CC_   768
#define H_    20
#define TIP_  4
#define D_    64
#define SCALE_ 0.125f   // 1/sqrt(64)

typedef __bf16 bf16;
typedef __attribute__((ext_vector_type(8)))  __bf16 bf16x8;
typedef __attribute__((ext_vector_type(16))) __bf16 bf16x16;
typedef __attribute__((ext_vector_type(8)))  float  floatx8;

#define SHUF16(lo, hi) __builtin_shufflevector(lo, hi, 0,1,2,3,4,5,6,7,8,9,10,11,12,13,14,15)

// ---------------- elementwise casts ----------------
__global__ __launch_bounds__(256) void cast_f32_bf16(const float* __restrict__ in,
                                                     bf16* __restrict__ out,
                                                     unsigned long long n) {
  unsigned long long i = (unsigned long long)blockIdx.x * blockDim.x + threadIdx.x;
  unsigned long long stride = (unsigned long long)gridDim.x * blockDim.x;
  for (; i < n; i += stride) out[i] = (bf16)in[i];
}

// w [K,N] f32 -> wt [N,K] bf16 (write-coalesced)
__global__ __launch_bounds__(256) void transpose_cast(const float* __restrict__ w,
                                                      bf16* __restrict__ wt,
                                                      int K, int N) {
  unsigned long long total = (unsigned long long)K * N;
  unsigned long long i = (unsigned long long)blockIdx.x * blockDim.x + threadIdx.x;
  unsigned long long stride = (unsigned long long)gridDim.x * blockDim.x;
  for (; i < total; i += stride) {
    int n = (int)(i / K);
    int k = (int)(i % K);
    wt[i] = (bf16)w[(unsigned long long)k * N + n];
  }
}

// ---------------- bf16 WMMA GEMM:  C[M,N] = A[M,K] * W[K,N] (+bias) ----------------
// A row-major bf16, Wt is W transposed: [N,K] bf16.
// One wave computes a 32(M) x 64(N) tile: 2 A fragments x 4 B fragments -> 8 WMMAs
// per 32-K step. K-loop is unrolled by 2 with two disjoint fragment register
// sets (ping/pong) so prefetched loads land directly in WMMA source registers.
// Requires K % 64 == 0.
__global__ __launch_bounds__(256)
void gemm_bf16_wmma(const bf16* __restrict__ A, const bf16* __restrict__ Wt,
                    const float* __restrict__ bias, float* __restrict__ Cb,
                    int M, int N, int K) {
  const int lane  = threadIdx.x & 31;
  const int wave  = threadIdx.x >> 5;
  const int c16   = lane & 15;   // col within 16-tile / A row within tile
  const int half  = lane >> 4;   // k-half selector
  const int n0    = blockIdx.x * 64;
  const int m0    = (blockIdx.y * 8 + wave) * 32;

  floatx8 acc0[4], acc1[4];
#pragma unroll
  for (int t = 0; t < 4; ++t) {
    acc0[t] = (floatx8){0.f,0.f,0.f,0.f,0.f,0.f,0.f,0.f};
    acc1[t] = (floatx8){0.f,0.f,0.f,0.f,0.f,0.f,0.f,0.f};
  }

  const bf16* __restrict__ arow0 = A + (unsigned long long)(m0 + c16) * K + half * 8;
  const bf16* __restrict__ arow1 = arow0 + (unsigned long long)16 * K;
  const bf16* __restrict__ brow[4];
#pragma unroll
  for (int t = 0; t < 4; ++t)
    brow[t] = Wt + (unsigned long long)(n0 + t * 16 + c16) * K + half * 8;

  // two fragment register sets
  bf16x8 pA0lo, pA0hi, pA1lo, pA1hi, pBl[4], pBh[4];   // ping
  bf16x8 qA0lo, qA0hi, qA1lo, qA1hi, qBl[4], qBh[4];   // pong

#define LOAD_SET(A0lo, A0hi, A1lo, A1hi, Bl, Bh, koff)                        \
  do {                                                                        \
    A0lo = *(const bf16x8*)(arow0 + (koff));                                  \
    A0hi = *(const bf16x8*)(arow0 + (koff) + 16);                             \
    A1lo = *(const bf16x8*)(arow1 + (koff));                                  \
    A1hi = *(const bf16x8*)(arow1 + (koff) + 16);                             \
    _Pragma("unroll")                                                         \
    for (int t = 0; t < 4; ++t) {                                             \
      Bl[t] = *(const bf16x8*)(brow[t] + (koff));                             \
      Bh[t] = *(const bf16x8*)(brow[t] + (koff) + 16);                        \
    }                                                                         \
  } while (0)

#define WMMA_SET(A0lo, A0hi, A1lo, A1hi, Bl, Bh)                              \
  do {                                                                        \
    bf16x16 a0_ = SHUF16(A0lo, A0hi);                                         \
    bf16x16 a1_ = SHUF16(A1lo, A1hi);                                         \
    _Pragma("unroll")                                                         \
    for (int t = 0; t < 4; ++t) {                                             \
      bf16x16 b_ = SHUF16(Bl[t], Bh[t]);                                      \
      acc0[t] = __builtin_amdgcn_wmma_f32_16x16x32_bf16(                      \
          false, a0_, false, b_, (short)0, acc0[t], false, false);            \
      acc1[t] = __builtin_amdgcn_wmma_f32_16x16x32_bf16(                      \
          false, a1_, false, b_, (short)0, acc1[t], false, false);            \
    }                                                                         \
  } while (0)

  // prologue: ping holds k = 0
  LOAD_SET(pA0lo, pA0hi, pA1lo, pA1hi, pBl, pBh, 0);

  const int nDouble = (K >> 6) - 1;   // K/64 - 1 double-steps
  int k = 32;
  for (int it = 0; it < nDouble; ++it) {
    LOAD_SET(qA0lo, qA0hi, qA1lo, qA1hi, qBl, qBh, k);        // pong <- k
    WMMA_SET(pA0lo, pA0hi, pA1lo, pA1hi, pBl, pBh);           // compute ping
    LOAD_SET(pA0lo, pA0hi, pA1lo, pA1hi, pBl, pBh, k + 32);   // ping <- k+32
    WMMA_SET(qA0lo, qA0hi, qA1lo, qA1hi, qBl, qBh);           // compute pong
    k += 64;
  }
  // tail: ping holds K-64's +32? No — ping holds K-32 after loop? Structure:
  // after loop, ping holds k-32 ... actually ping holds (k - 32) = K - 64.
  // Remaining offsets: K-64 computed? Walkthrough: offsets consumed so far:
  // 0,32,...,k-32 with k = 32 + 64*nDouble = K - 32. Remaining: K-32 (loaded into pong); ping holds K-64.
  LOAD_SET(qA0lo, qA0hi, qA1lo, qA1hi, qBl, qBh, k);          // pong <- K-32
  WMMA_SET(pA0lo, pA0hi, pA1lo, pA1hi, pBl, pBh);             // K-64
  WMMA_SET(qA0lo, qA0hi, qA1lo, qA1hi, qBl, qBh);             // K-32

#undef LOAD_SET
#undef WMMA_SET

#pragma unroll
  for (int t = 0; t < 4; ++t) {
    const int cc = n0 + t * 16 + c16;
    const float bval = bias ? bias[cc] : 0.f;
#pragma unroll
    for (int r = 0; r < 8; ++r) {
      const int rr = m0 + half * 8 + r;
      Cb[(unsigned long long)rr * N + cc] = acc0[t][r] + bval;
      Cb[(unsigned long long)(rr + 16) * N + cc] = acc1[t][r] + bval;
    }
  }
}

// ---------------- K/V (+IP K/V) projection: ehs[row,CC] @ W[CC,C] ----------------
__global__ __launch_bounds__(256)
void kv_proj(const float* __restrict__ ehs, const float* __restrict__ iph,
             const float* __restrict__ wk,  const float* __restrict__ wv,
             const float* __restrict__ wkip, const float* __restrict__ wvip,
             float* __restrict__ kbuf, float* __restrict__ vbuf,
             float* __restrict__ ipk,  float* __restrict__ ipv) {
  __shared__ float xrow[CC_];
  const int r = blockIdx.x;
  const float* src; float* dk; float* dv; const float* wkp; const float* wvp;
  if (r < B_ * STXT_) {
    src = ehs + (unsigned long long)r * CC_;
    dk = kbuf + (unsigned long long)r * C_;
    dv = vbuf + (unsigned long long)r * C_;
    wkp = wk; wvp = wv;
  } else {
    const int t = r - B_ * STXT_;
    src = iph + (unsigned long long)t * CC_;
    dk = ipk + (unsigned long long)t * C_;
    dv = ipv + (unsigned long long)t * C_;
    wkp = wkip; wvp = wvip;
  }
  for (int i = threadIdx.x; i < CC_; i += 256) xrow[i] = src[i];
  __syncthreads();
  for (int c = threadIdx.x; c < C_; c += 256) {
    float sk = 0.f, sv = 0.f;
    for (int m = 0; m < CC_; ++m) {
      const float x = xrow[m];
      sk = fmaf(x, wkp[(unsigned long long)m * C_ + c], sk);
      sv = fmaf(x, wvp[(unsigned long long)m * C_ + c], sv);
    }
    dk[c] = sk; dv[c] = sv;
  }
}

// ---------------- pass 1: deterministic per-block (sum, sumsq) of scores ----------------
__global__ __launch_bounds__(256)
void score_stats(const float* __restrict__ q, const float* __restrict__ kbuf,
                 double* __restrict__ partials) {
  __shared__ float kt[STXT_ * D_];
  __shared__ double rsum[256];
  __shared__ double rsq[256];
  const int b = blockIdx.z, h = blockIdx.y;
  const int s = blockIdx.x * 256 + threadIdx.x;
  for (int i = threadIdx.x; i < STXT_ * D_; i += 256) {
    const int j = i >> 6, d = i & 63;
    kt[i] = kbuf[((unsigned long long)(b * STXT_ + j)) * C_ + h * D_ + d];
  }
  __syncthreads();

  const float4* __restrict__ qp =
      (const float4*)(q + ((unsigned long long)b * S_ + s) * C_ + h * D_);
  float4 qv[16];
#pragma unroll
  for (int i = 0; i < 16; ++i) qv[i] = qp[i];

  double lsum = 0.0, lsq = 0.0;
  for (int j = 0; j < STXT_; ++j) {
    const float4* kr = (const float4*)(kt + j * D_);
    float dot = 0.f;
#pragma unroll
    for (int i = 0; i < 16; ++i) {
      float4 kk = kr[i];
      dot = fmaf(qv[i].x, kk.x, dot);
      dot = fmaf(qv[i].y, kk.y, dot);
      dot = fmaf(qv[i].z, kk.z, dot);
      dot = fmaf(qv[i].w, kk.w, dot);
    }
    const float sc = dot * SCALE_;
    lsum += (double)sc;
    lsq  += (double)sc * (double)sc;
  }
  rsum[threadIdx.x] = lsum;
  rsq[threadIdx.x]  = lsq;
  __syncthreads();
  for (int off = 128; off > 0; off >>= 1) {
    if ((int)threadIdx.x < off) {
      rsum[threadIdx.x] += rsum[threadIdx.x + off];
      rsq[threadIdx.x]  += rsq[threadIdx.x + off];
    }
    __syncthreads();
  }
  if (threadIdx.x == 0) {
    const int bid = (b * gridDim.y + h) * gridDim.x + blockIdx.x;
    partials[2 * bid]     = rsum[0];
    partials[2 * bid + 1] = rsq[0];
  }
}

__global__ __launch_bounds__(256)
void finalize_stats(const double* __restrict__ partials, int nblocks,
                    const float* __restrict__ sigma, float* __restrict__ bias_scale) {
  __shared__ double rs[256];
  __shared__ double rq[256];
  double s = 0.0, q2 = 0.0;
  for (int i = threadIdx.x; i < nblocks; i += 256) {
    s  += partials[2 * i];
    q2 += partials[2 * i + 1];
  }
  rs[threadIdx.x] = s; rq[threadIdx.x] = q2;
  __syncthreads();
  for (int off = 128; off > 0; off >>= 1) {
    if ((int)threadIdx.x < off) {
      rs[threadIdx.x] += rs[threadIdx.x + off];
      rq[threadIdx.x] += rq[threadIdx.x + off];
    }
    __syncthreads();
  }
  if (threadIdx.x == 0) {
    const double N = (double)B_ * H_ * S_ * STXT_;
    const double var = (rq[0] - rs[0] * rs[0] / N) / (N - 1.0);
    *bias_scale = (float)((double)sigma[0] * sqrt(var > 0.0 ? var : 0.0));
  }
}

// ---------------- pass 2: fused biased softmax + PV + IP stream -> h (bf16) ----------------
__global__ __launch_bounds__(256)
void attn_fused(const float* __restrict__ q, const float* __restrict__ kbuf,
                const float* __restrict__ vbuf, const float* __restrict__ ipk,
                const float* __restrict__ ipv, const float* __restrict__ region,
                const float* __restrict__ bias_scale_p, bf16* __restrict__ hout) {
  __shared__ float kt[STXT_ * D_];
  __shared__ float vt[STXT_ * D_];
  __shared__ float ipkt[TIP_ * D_];
  __shared__ float ipvt[TIP_ * D_];
  const int b = blockIdx.z, h = blockIdx.y;
  const int s = blockIdx.x * 256 + threadIdx.x;
  for (int i = threadIdx.x; i < STXT_ * D_; i += 256) {
    const int j = i >> 6, d = i & 63;
    const unsigned long long src = ((unsigned long long)(b * STXT_ + j)) * C_ + h * D_ + d;
    kt[i] = kbuf[src];
    vt[i] = vbuf[src];
  }
  for (int i = threadIdx.x; i < TIP_ * D_; i += 256) {
    const int j = i >> 6, d = i & 63;
    const unsigned long long src = ((unsigned long long)(b * TIP_ + j)) * C_ + h * D_ + d;
    ipkt[i] = ipk[src];
    ipvt[i] = ipv[src];
  }
  __syncthreads();

  const float bscale = *bias_scale_p;
  const float4* __restrict__ qp =
      (const float4*)(q + ((unsigned long long)b * S_ + s) * C_ + h * D_);
  float4 qv[16];
#pragma unroll
  for (int i = 0; i < 16; ++i) qv[i] = qp[i];
  const float* __restrict__ rrow = region + ((unsigned long long)b * S_ + s) * STXT_;

  // pass 1: running max
  float mx = -INFINITY;
  for (int j = 0; j < STXT_; ++j) {
    const float4* kr = (const float4*)(kt + j * D_);
    float dot = 0.f;
#pragma unroll
    for (int i = 0; i < 16; ++i) {
      float4 kk = kr[i];
      dot = fmaf(qv[i].x, kk.x, dot);
      dot = fmaf(qv[i].y, kk.y, dot);
      dot = fmaf(qv[i].z, kk.z, dot);
      dot = fmaf(qv[i].w, kk.w, dot);
    }
    mx = fmaxf(mx, fmaf(rrow[j], bscale, dot * SCALE_));
  }

  // pass 2: exp + PV accumulate
  float denom = 0.f;
  float4 acc[16];
#pragma unroll
  for (int i = 0; i < 16; ++i) acc[i] = make_float4(0.f, 0.f, 0.f, 0.f);
  for (int j = 0; j < STXT_; ++j) {
    const float4* kr = (const float4*)(kt + j * D_);
    float dot = 0.f;
#pragma unroll
    for (int i = 0; i < 16; ++i) {
      float4 kk = kr[i];
      dot = fmaf(qv[i].x, kk.x, dot);
      dot = fmaf(qv[i].y, kk.y, dot);
      dot = fmaf(qv[i].z, kk.z, dot);
      dot = fmaf(qv[i].w, kk.w, dot);
    }
    const float e = __expf(fmaf(rrow[j], bscale, dot * SCALE_) - mx);
    denom += e;
    const float4* vr = (const float4*)(vt + j * D_);
#pragma unroll
    for (int i = 0; i < 16; ++i) {
      float4 vv = vr[i];
      acc[i].x = fmaf(e, vv.x, acc[i].x);
      acc[i].y = fmaf(e, vv.y, acc[i].y);
      acc[i].z = fmaf(e, vv.z, acc[i].z);
      acc[i].w = fmaf(e, vv.w, acc[i].w);
    }
  }

  // IP stream: 4 keys, plain softmax
  float ipsc[TIP_];
  float ipmx = -INFINITY;
#pragma unroll
  for (int t = 0; t < TIP_; ++t) {
    const float4* kr = (const float4*)(ipkt + t * D_);
    float dot = 0.f;
#pragma unroll
    for (int i = 0; i < 16; ++i) {
      float4 kk = kr[i];
      dot = fmaf(qv[i].x, kk.x, dot);
      dot = fmaf(qv[i].y, kk.y, dot);
      dot = fmaf(qv[i].z, kk.z, dot);
      dot = fmaf(qv[i].w, kk.w, dot);
    }
    ipsc[t] = dot * SCALE_;
    ipmx = fmaxf(ipmx, ipsc[t]);
  }
  float ipden = 0.f;
#pragma unroll
  for (int t = 0; t < TIP_; ++t) { ipsc[t] = __expf(ipsc[t] - ipmx); ipden += ipsc[t]; }

  const float inv = 1.f / denom;
  const float ipinv = 1.f / ipden;
  bf16* __restrict__ out = hout + ((unsigned long long)b * S_ + s) * C_ + h * D_;
#pragma unroll
  for (int i = 0; i < 16; ++i) {
    const int d = i * 4;
    float o0 = acc[i].x * inv, o1 = acc[i].y * inv, o2 = acc[i].z * inv, o3 = acc[i].w * inv;
#pragma unroll
    for (int t = 0; t < TIP_; ++t) {
      const float p = ipsc[t] * ipinv;
      o0 = fmaf(p, ipvt[t * D_ + d + 0], o0);
      o1 = fmaf(p, ipvt[t * D_ + d + 1], o1);
      o2 = fmaf(p, ipvt[t * D_ + d + 2], o2);
      o3 = fmaf(p, ipvt[t * D_ + d + 3], o3);
    }
    out[d + 0] = (bf16)o0; out[d + 1] = (bf16)o1;
    out[d + 2] = (bf16)o2; out[d + 3] = (bf16)o3;
  }
}

// ---------------- launch ----------------
extern "C" void kernel_launch(void* const* d_in, const int* in_sizes, int n_in,
                              void* d_out, int out_size, void* d_ws, size_t ws_size,
                              hipStream_t stream) {
  const float* hs    = (const float*)d_in[0];   // [B,S,C]
  const float* ehs   = (const float*)d_in[1];   // [B,Stxt,CC]
  const float* iph   = (const float*)d_in[2];   // [B,Tip,CC]
  const float* rgn   = (const float*)d_in[3];   // [B,S,Stxt]
  const float* sigma = (const float*)d_in[4];   // scalar
  const float* w_q   = (const float*)d_in[5];   // [C,C]
  const float* w_k   = (const float*)d_in[6];   // [CC,C]
  const float* w_v   = (const float*)d_in[7];
  const float* w_kip = (const float*)d_in[8];
  const float* w_vip = (const float*)d_in[9];
  const float* w_out = (const float*)d_in[10];  // [C,C]
  const float* b_out = (const float*)d_in[11];  // [C]
  float* out = (float*)d_out;

  // workspace carve-up (256B aligned), total ~130 MB
  size_t off = 0;
  char* ws = (char*)d_ws;
  auto carve = [&](size_t bytes) -> void* {
    void* p = ws + off;
    off = (off + bytes + 255) & ~(size_t)255;
    return p;
  };
  const size_t nBSC = (size_t)B_ * S_ * C_;
  float*  q_f32   = (float*)carve(nBSC * sizeof(float));
  bf16*   hs_bf16 = (bf16*)carve(nBSC * sizeof(bf16));   // reused as h after attention
  bf16*   wqt     = (bf16*)carve((size_t)C_ * C_ * sizeof(bf16));
  bf16*   wot     = (bf16*)carve((size_t)C_ * C_ * sizeof(bf16));
  float*  kbuf    = (float*)carve((size_t)B_ * STXT_ * C_ * sizeof(float));
  float*  vbuf    = (float*)carve((size_t)B_ * STXT_ * C_ * sizeof(float));
  float*  ipk     = (float*)carve((size_t)B_ * TIP_ * C_ * sizeof(float));
  float*  ipv     = (float*)carve((size_t)B_ * TIP_ * C_ * sizeof(float));
  const int nStatBlocks = (S_ / 256) * H_ * B_;          // 1280
  double* partials = (double*)carve((size_t)nStatBlocks * 2 * sizeof(double));
  float*  bscale   = (float*)carve(256);
  (void)ws_size; (void)in_sizes; (void)n_in; (void)out_size;

  // 1. cast hidden_states to bf16
  cast_f32_bf16<<<4096, 256, 0, stream>>>(hs, hs_bf16, nBSC);
  // 2. transpose+cast the two 1280x1280 weights
  transpose_cast<<<2048, 256, 0, stream>>>(w_q, wqt, C_, C_);
  transpose_cast<<<2048, 256, 0, stream>>>(w_out, wot, C_, C_);
  // 3. K/V and IP-K/V projections (f32)
  kv_proj<<<B_ * (STXT_ + TIP_), 256, 0, stream>>>(ehs, iph, w_k, w_v, w_kip, w_vip,
                                                   kbuf, vbuf, ipk, ipv);
  // 4. Q = hs @ w_q  (bf16 WMMA, f32 out)
  {
    dim3 grid(C_ / 64, (B_ * S_) / 256);
    gemm_bf16_wmma<<<grid, 256, 0, stream>>>(hs_bf16, wqt, nullptr, q_f32,
                                             B_ * S_, C_, C_);
  }
  // 5. global score statistics (deterministic two-stage reduction)
  {
    dim3 grid(S_ / 256, H_, B_);
    score_stats<<<grid, 256, 0, stream>>>(q_f32, kbuf, partials);
  }
  finalize_stats<<<1, 256, 0, stream>>>(partials, nStatBlocks, sigma, bscale);
  // 6. fused biased-softmax attention + IP stream -> h (bf16, aliases hs_bf16)
  {
    dim3 grid(S_ / 256, H_, B_);
    attn_fused<<<grid, 256, 0, stream>>>(q_f32, kbuf, vbuf, ipk, ipv, rgn,
                                         bscale, hs_bf16);
  }
  // 7. out = h @ w_out + b_out  (bf16 WMMA, f32 out)
  {
    dim3 grid(C_ / 64, (B_ * S_) / 256);
    gemm_bf16_wmma<<<grid, 256, 0, stream>>>(hs_bf16, wot, b_out, out,
                                             B_ * S_, C_, C_);
  }
}
